// MultiHeadAttention_20134806683804
// MI455X (gfx1250) — matrix operand order repaired and checked by reading, not compiled, not run on hardware
//
#include <hip/hip_runtime.h>

typedef __attribute__((ext_vector_type(16))) __bf16 v16bf;
typedef __attribute__((ext_vector_type(8)))  __bf16 bf16x8;
typedef __attribute__((ext_vector_type(8)))  float  v8f;

constexpr int Bn = 128, Tn = 256, Dn = 512, Hn = 8, HDn = 64;

__device__ __forceinline__ v8f wmma_bf16(v16bf a, v16bf b, v8f c) {
  // D = A(16x32 bf16) * B(32x16 bf16) + C(16x16 f32)
  return __builtin_amdgcn_wmma_f32_16x16x32_bf16(false, a, false, b, (short)0, c,
                                                 false, false);
}

// A-fragment (16x32, 16-bit): lane holds row m=lane&15; elements 0..15 -> K=2*kb0+0..15,
// where kb0 = (lane<16 ? 0 : 8).
__device__ __forceinline__ v16bf ldfragA(const __bf16* row, int kb0) {
  union { v16bf v; bf16x8 h[2]; } u;
  u.h[0] = *(const bf16x8*)(row + 2 * kb0);
  u.h[1] = *(const bf16x8*)(row + 2 * kb0 + 8);
  return u.v;
}

// Row reductions across the 16 lanes that share a C-matrix row (wave32, width-16 groups).
__device__ __forceinline__ float redmax16(float v) {
  v = fmaxf(v, __shfl_xor(v, 8, 16));
  v = fmaxf(v, __shfl_xor(v, 4, 16));
  v = fmaxf(v, __shfl_xor(v, 2, 16));
  v = fmaxf(v, __shfl_xor(v, 1, 16));
  return v;
}
__device__ __forceinline__ float redsum16(float v) {
  v += __shfl_xor(v, 8, 16);
  v += __shfl_xor(v, 4, 16);
  v += __shfl_xor(v, 2, 16);
  v += __shfl_xor(v, 1, 16);
  return v;
}

// ---------------- conversion / transpose ----------------

__global__ void cvt_bf16_kernel(const float* __restrict__ src, __bf16* __restrict__ dst, int n) {
  int i = blockIdx.x * blockDim.x + threadIdx.x;
  if (i < n) dst[i] = (__bf16)src[i];
}

// src: [batch][rows][cols] f32 -> dst: [batch][cols][rows] bf16 (batch = blockIdx.y)
__global__ void transpose_bf16_kernel(const float* __restrict__ src, __bf16* __restrict__ dst,
                                      int rows, int cols) {
  int i = blockIdx.x * blockDim.x + threadIdx.x;
  int total = rows * cols;
  if (i >= total) return;
  int c = i / rows, r = i % rows;
  size_t base = (size_t)blockIdx.y * total;
  dst[base + i] = (__bf16)src[base + (size_t)r * cols + c];
}

// ---------------- QKV projection: q,k [B,H,T,HD]; v stored transposed [B,H,HD,T] ----------------

__global__ __launch_bounds__(32) void qkv_proj_kernel(
    const __bf16* __restrict__ xb,
    const __bf16* __restrict__ wqT, const __bf16* __restrict__ wkT,
    const __bf16* __restrict__ wvT,
    __bf16* __restrict__ qb, __bf16* __restrict__ kbuf, __bf16* __restrict__ vtb) {
  const int lane = threadIdx.x;
  const int bid  = blockIdx.x;
  const int tt = bid & 15, h = (bid >> 4) & 7, b = bid >> 7;
  const int t0 = tt * 16;
  const int nrow = lane & 15;
  const int hi   = lane >> 4;
  const int kb0  = hi * 8;    // A-fragment K base
  const int kBb  = hi * 16;   // B-fragment K base (lanes 16-31 hold K=16..31)

  const __bf16* xrow = xb + ((size_t)(b * Tn + t0 + nrow)) * Dn;
  const __bf16* wq = wqT + ((size_t)h * HDn + nrow) * Dn + kBb;
  const __bf16* wk = wkT + ((size_t)h * HDn + nrow) * Dn + kBb;
  const __bf16* wv = wvT + ((size_t)h * HDn + nrow) * Dn + kBb;

  v8f qa[4] = {}, ka[4] = {}, va[4] = {};
#pragma unroll 4
  for (int kk = 0; kk < Dn / 32; ++kk) {
    const int ko = kk * 32;
    v16bf af = ldfragA(xrow + ko, kb0);
#pragma unroll
    for (int nt = 0; nt < 4; ++nt) {
      v16bf bq = *(const v16bf*)(wq + (size_t)nt * 16 * Dn + ko);
      qa[nt] = wmma_bf16(af, bq, qa[nt]);
      v16bf bk = *(const v16bf*)(wk + (size_t)nt * 16 * Dn + ko);
      ka[nt] = wmma_bf16(af, bk, ka[nt]);
      v16bf bv = *(const v16bf*)(wv + (size_t)nt * 16 * Dn + ko);
      va[nt] = wmma_bf16(af, bv, va[nt]);
    }
  }
  const size_t bh = (size_t)(b * Hn + h);
#pragma unroll
  for (int r = 0; r < 8; ++r) {
    const int t = t0 + r + hi * 8;  // C layout: VGPR r -> row r (lanes 0-15) / r+8 (lanes 16-31)
#pragma unroll
    for (int nt = 0; nt < 4; ++nt) {
      const int hd = nt * 16 + nrow;
      qb  [(bh * Tn + t) * HDn + hd] = (__bf16)qa[nt][r];
      kbuf[(bh * Tn + t) * HDn + hd] = (__bf16)ka[nt][r];
      vtb [(bh * HDn + hd) * Tn + t] = (__bf16)va[nt][r];
    }
  }
}

// ---------------- causal flash attention core, one wave = 16 query rows ----------------

__global__ __launch_bounds__(32) void attn_core_kernel(
    const __bf16* __restrict__ qb, const __bf16* __restrict__ kbuf,
    const __bf16* __restrict__ vtb, __bf16* __restrict__ aout) {
  __shared__ __align__(32) __bf16 pshm[16 * 32];
  const int lane = threadIdx.x;
  const int bid  = blockIdx.x;
  const int tt = bid & 15, h = (bid >> 4) & 7, b = bid >> 7;
  const int t0 = tt * 16;
  const int nrow = lane & 15;
  const int hi   = lane >> 4;
  const int kb0  = hi * 8;
  const int kBb  = hi * 16;
  const float scale = 0.125f;  // HD^-0.5

  const size_t bh = (size_t)(b * Hn + h);
  const __bf16* qrow = qb + (bh * Tn + t0 + nrow) * HDn;
  v16bf qf0 = ldfragA(qrow, kb0);        // K = 0..31 of head dim
  v16bf qf1 = ldfragA(qrow + 32, kb0);   // K = 32..63

  float mrow[8], lrow[8];
  v8f oacc[4] = {};
#pragma unroll
  for (int r = 0; r < 8; ++r) { mrow[r] = -1e30f; lrow[r] = 0.0f; }

  const int t_hi = t0 + 15;
  for (int s0 = 0; s0 <= t_hi; s0 += 32) {
    // scores tile 16x32: two 16x16 WMMA accumulations over K=64
    const __bf16* kcol0 = kbuf + (bh * Tn + s0 + nrow) * HDn + kBb;
    const __bf16* kcol1 = kcol0 + 16 * HDn;
    v16bf kf00 = *(const v16bf*)(kcol0);
    v16bf kf01 = *(const v16bf*)(kcol0 + 32);
    v16bf kf10 = *(const v16bf*)(kcol1);
    v16bf kf11 = *(const v16bf*)(kcol1 + 32);
    v8f sa = {}, sb = {};
    sa = wmma_bf16(qf0, kf00, sa);
    sa = wmma_bf16(qf1, kf01, sa);
    sb = wmma_bf16(qf0, kf10, sb);
    sb = wmma_bf16(qf1, kf11, sb);

    // online softmax update per owned row
#pragma unroll
    for (int r = 0; r < 8; ++r) {
      const int t = t0 + r + hi * 8;
      float a = (s0 + nrow      <= t) ? sa[r] * scale : -1e30f;
      float c = (s0 + 16 + nrow <= t) ? sb[r] * scale : -1e30f;
      float rm  = redmax16(fmaxf(a, c));
      float mn  = fmaxf(mrow[r], rm);
      float corr = __expf(mrow[r] - mn);
      float p0 = __expf(a - mn);
      float p1 = __expf(c - mn);
      lrow[r] = lrow[r] * corr + redsum16(p0 + p1);
      mrow[r] = mn;
#pragma unroll
      for (int nt = 0; nt < 4; ++nt) oacc[nt][r] *= corr;
      const int m = r + hi * 8;
      pshm[m * 32 + nrow]      = (__bf16)p0;
      pshm[m * 32 + 16 + nrow] = (__bf16)p1;
    }
    __syncthreads();
    // P (16x32) as A-fragment via LDS; V^T rows give contiguous B-fragments
    v16bf pf = ldfragA(&pshm[nrow * 32], kb0);
    const __bf16* vcol = vtb + (bh * HDn + nrow) * Tn + s0 + kBb;
#pragma unroll
    for (int nt = 0; nt < 4; ++nt) {
      v16bf vf = *(const v16bf*)(vcol + (size_t)nt * 16 * Tn);
      oacc[nt] = wmma_bf16(pf, vf, oacc[nt]);
    }
    __syncthreads();
  }

#pragma unroll
  for (int r = 0; r < 8; ++r) {
    const int t = t0 + r + hi * 8;
    const float inv = 1.0f / lrow[r];
#pragma unroll
    for (int nt = 0; nt < 4; ++nt) {
      aout[((size_t)b * Tn + t) * Dn + h * HDn + nt * 16 + nrow] =
          (__bf16)(oacc[nt][r] * inv);
    }
  }
}

// ---------------- output projection: y = attn_out @ Wo + bo ----------------

__global__ __launch_bounds__(32) void out_proj_kernel(
    const __bf16* __restrict__ aout, const __bf16* __restrict__ woT,
    const float* __restrict__ bo, float* __restrict__ y) {
  const int lane = threadIdx.x;
  const int bid  = blockIdx.x;
  const int ct = bid & 7;      // 64-wide column group
  const int mt = bid >> 3;     // 16-row tile over B*T
  const int t0 = mt * 16;
  const int nrow = lane & 15;
  const int hi   = lane >> 4;
  const int kb0  = hi * 8;
  const int kBb  = hi * 16;

  const __bf16* arow = aout + ((size_t)t0 + nrow) * Dn;
  const __bf16* wrow = woT + ((size_t)(ct * 64 + nrow)) * Dn + kBb;
  v8f acc[4] = {};
#pragma unroll 4
  for (int kk = 0; kk < Dn / 32; ++kk) {
    const int ko = kk * 32;
    v16bf af = ldfragA(arow + ko, kb0);
#pragma unroll
    for (int nt = 0; nt < 4; ++nt) {
      v16bf bw = *(const v16bf*)(wrow + (size_t)nt * 16 * Dn + ko);
      acc[nt] = wmma_bf16(af, bw, acc[nt]);
    }
  }
#pragma unroll
  for (int r = 0; r < 8; ++r) {
    const int row = t0 + r + hi * 8;
#pragma unroll
    for (int nt = 0; nt < 4; ++nt) {
      const int n = ct * 64 + nt * 16 + nrow;
      y[(size_t)row * Dn + n] = acc[nt][r] + bo[n];
    }
  }
}

// ---------------- launcher ----------------

extern "C" void kernel_launch(void* const* d_in, const int* in_sizes, int n_in,
                              void* d_out, int out_size, void* d_ws, size_t ws_size,
                              hipStream_t stream) {
  (void)in_sizes; (void)n_in; (void)out_size; (void)ws_size;
  const float* x  = (const float*)d_in[0];
  const float* Wq = (const float*)d_in[1];
  const float* Wk = (const float*)d_in[2];
  const float* Wv = (const float*)d_in[3];
  const float* Wo = (const float*)d_in[4];
  const float* bo = (const float*)d_in[5];
  float* y = (float*)d_out;

  char* ws = (char*)d_ws;
  size_t off = 0;
  auto carve = [&](size_t bytes) {
    void* p = ws + off;
    off += (bytes + 255) & ~(size_t)255;
    return p;
  };
  __bf16* xb  = (__bf16*)carve((size_t)Bn * Tn * Dn * 2);
  __bf16* wqT = (__bf16*)carve((size_t)Hn * HDn * Dn * 2);
  __bf16* wkT = (__bf16*)carve((size_t)Hn * HDn * Dn * 2);
  __bf16* wvT = (__bf16*)carve((size_t)Hn * HDn * Dn * 2);
  __bf16* woT = (__bf16*)carve((size_t)Dn * Dn * 2);
  __bf16* qb  = (__bf16*)carve((size_t)Bn * Hn * Tn * HDn * 2);
  __bf16* kb  = (__bf16*)carve((size_t)Bn * Hn * Tn * HDn * 2);
  __bf16* vtb = (__bf16*)carve((size_t)Bn * Hn * HDn * Tn * 2);
  __bf16* ao  = (__bf16*)carve((size_t)Bn * Tn * Dn * 2);

  {
    int n = Bn * Tn * Dn;
    cvt_bf16_kernel<<<(n + 255) / 256, 256, 0, stream>>>(x, xb, n);
  }
  {
    dim3 g((Dn * HDn + 255) / 256, Hn);
    transpose_bf16_kernel<<<g, 256, 0, stream>>>(Wq, wqT, Dn, HDn);
    transpose_bf16_kernel<<<g, 256, 0, stream>>>(Wk, wkT, Dn, HDn);
    transpose_bf16_kernel<<<g, 256, 0, stream>>>(Wv, wvT, Dn, HDn);
  }
  {
    dim3 g((Dn * Dn + 255) / 256, 1);
    transpose_bf16_kernel<<<g, 256, 0, stream>>>(Wo, woT, Dn, Dn);
  }

  const int tiles = Bn * Hn * (Tn / 16);  // 16384 waves
  qkv_proj_kernel<<<tiles, 32, 0, stream>>>(xb, wqT, wkT, wvT, qb, kb, vtb);
  attn_core_kernel<<<tiles, 32, 0, stream>>>(qb, kb, vtb, ao);
  out_proj_kernel<<<(Bn * Tn / 16) * (Dn / 64), 32, 0, stream>>>(ao, woT, bo, y);
}
